// VarianceAdaptor_65343632441697
// MI455X (gfx1250) — compile-verified, matrix operand order checked by reading, not run on hardware
//
#include <hip/hip_runtime.h>
#include <hip/hip_bf16.h>

// ---------------------------------------------------------------------------
// VarianceAdaptor for MI455X (gfx1250, wave32, WMMA).
//
// Conv1d(K=3,256->256) == GEMM [16 x 768] x [768 x 256] per 16-row tile,
// run on v_wmma_f32_16x16x32_bf16 with f32 emulated by bf16 hi/lo split:
//   A*B ~= Ahi*Bhi + Ahi*Blo + Alo*Bhi   (error ~2^-22).
// Weights are pre-split & pre-swizzled into per-lane fragment order once,
// activations are split at LDS stage-in, so the hot loop is only
// b128 fragment loads + 6 WMMAs per K-step.
// ---------------------------------------------------------------------------

static constexpr int Hc  = 256;
static constexpr int FSc = 256;
static constexpr int Bc  = 64;
static constexpr int Lc  = 512;
static constexpr int Tc  = 2048;
static constexpr float EPSc = 1e-5f;

typedef __attribute__((ext_vector_type(16))) __bf16 v16bf;
typedef __attribute__((ext_vector_type(8)))  __bf16 v8bf;
typedef __attribute__((ext_vector_type(8)))  float  v8f;

// pack sizes: per conv layer, 24 K-steps x 16 N-tiles x 32 lanes x 16 bf16
static constexpr size_t PACK_ELEMS = 24ull * 16 * 32 * 16;   // 196608 bf16
static constexpr size_t H1_BYTES   = (size_t)Bc * Lc * 256 * sizeof(float);
static constexpr size_t CUM_BYTES  = (size_t)Bc * Lc * sizeof(int);

// round-to-nearest-even f32 -> bf16 (bit form), and back
static __device__ __forceinline__ unsigned short bf16_rne(float f) {
    unsigned u = __float_as_uint(f);
    unsigned r = u + 0x7FFFu + ((u >> 16) & 1u);
    return (unsigned short)(r >> 16);
}
static __device__ __forceinline__ float bf16_f32(unsigned short h) {
    return __uint_as_float(((unsigned)h) << 16);
}

// K_local(e) for the 16-bit A/B fragment layout (ISA 7.12.2):
// lanes 0-15 hold K {0..7,16..23}, lanes 16-31 hold K {8..15,24..31},
// VGPR r holds the pair (2r, 2r+1) within its half.
static __device__ __forceinline__ int frag_klocal(int e, int g) {
    int r = e >> 1, p = e & 1;
    return (r < 4) ? (g * 8 + (r << 1) + p)
                   : (16 + g * 8 + ((r - 4) << 1) + p);
}

// ---------------------------------------------------------------------------
// one-time weight split+swizzle: w (768,256) f32 -> hi/lo bf16 in per-lane
// fragment order: [(kk*16 + ntile)*32 + lane]*16 + e   (32B contiguous/lane)
// ---------------------------------------------------------------------------
__global__ __launch_bounds__(32)
void pack_weights_kernel(const float* __restrict__ w,
                         __bf16* __restrict__ hi,
                         __bf16* __restrict__ lo)
{
    const int kk   = blockIdx.x;      // 0..23
    const int nt16 = blockIdx.y;      // 0..15
    const int lane = threadIdx.x;     // 0..31
    const int g    = lane >> 4;
    const int nn   = lane & 15;
    const size_t base = (((size_t)kk * 16 + nt16) * 32 + lane) * 16;
    #pragma unroll
    for (int e = 0; e < 16; ++e) {
        int k = kk * 32 + frag_klocal(e, g);
        float v = w[(size_t)k * 256 + nt16 * 16 + nn];
        unsigned short hu = bf16_rne(v);
        hi[base + e] = __builtin_bit_cast(__bf16, hu);
        lo[base + e] = __builtin_bit_cast(__bf16, bf16_rne(v - bf16_f32(hu)));
    }
}

// ---------------------------------------------------------------------------
// Fused conv1d(K=3) + bias + ReLU + LayerNorm  [+ optional FS->1 linear]
// mode 0: write normalized h tile (16x256) to hout (workspace)
// mode 1: apply linear lw,lb + src_mask, write scalar per row to pred
// ---------------------------------------------------------------------------
__global__ __launch_bounds__(256)
void conv_ln_wmma_kernel(const float* __restrict__ in,    // (B,L,256)
                         const __bf16* __restrict__ whi,  // packed hi weights
                         const __bf16* __restrict__ wlo,  // packed lo weights
                         const float* __restrict__ bias,  // (256)
                         const float* __restrict__ gam,   // (256)
                         const float* __restrict__ bet,   // (256)
                         float* __restrict__ hout,        // (B,L,256) or nullptr
                         const float* __restrict__ lw,    // (256) or nullptr
                         const float* __restrict__ lb,    // (1)   or nullptr
                         const unsigned char* __restrict__ smask, // (B,L)
                         float* __restrict__ pred,        // (B,L) or nullptr
                         int mode)
{
    __shared__ __attribute__((aligned(16))) __bf16 sHi[18 * 256]; // halo tile hi
    __shared__ __attribute__((aligned(16))) __bf16 sLo[18 * 256]; // halo tile lo
    __shared__ float sH[16 * 256];                                // conv out tile
    __shared__ float sRed[2][16][16];
    __shared__ float sMu[16], sRs[16];

    const int bid = blockIdx.x;
    const int b   = bid / (Lc / 16);
    const int r0  = (bid % (Lc / 16)) * 16;
    const float* inB = in + (size_t)b * Lc * 256;

    // stage input rows (zero padded at sequence edges), split to hi/lo bf16
    for (int i = threadIdx.x; i < 18 * 256; i += 256) {
        int rw = i >> 8, col = i & 255;
        int l = r0 - 1 + rw;
        float v = (l >= 0 && l < Lc) ? inB[l * 256 + col] : 0.0f;
        unsigned short hu = bf16_rne(v);
        sHi[i] = __builtin_bit_cast(__bf16, hu);
        sLo[i] = __builtin_bit_cast(__bf16, bf16_rne(v - bf16_f32(hu)));
    }
    __syncthreads();

    const int wave = threadIdx.x >> 5;
    const int lane = threadIdx.x & 31;
    const int g    = lane >> 4;    // K-half selector
    const int am   = lane & 15;    // A row (M) / C col (N) within 16

    v8f acc[2];
    acc[0] = (v8f)0.0f;
    acc[1] = (v8f)0.0f;

    // GEMM K = 768 = 3 taps * 256 ch; each K-step of 32 stays in one tap
    #pragma unroll 1
    for (int kk = 0; kk < 24; ++kk) {
        const int kt  = kk >> 3;             // tap 0..2
        const int cbc = (kk << 5) & 255;     // channel base in tap
        const int rb  = (am + kt) * 256 + cbc + g * 8;

        v8bf h0 = *(const v8bf*)(&sHi[rb]);
        v8bf h1 = *(const v8bf*)(&sHi[rb + 16]);
        v8bf l0 = *(const v8bf*)(&sLo[rb]);
        v8bf l1 = *(const v8bf*)(&sLo[rb + 16]);
        v16bf ahi = __builtin_shufflevector(h0, h1, 0,1,2,3,4,5,6,7,8,9,10,11,12,13,14,15);
        v16bf alo = __builtin_shufflevector(l0, l1, 0,1,2,3,4,5,6,7,8,9,10,11,12,13,14,15);

        #pragma unroll
        for (int nt = 0; nt < 2; ++nt) {
            size_t foff = (((size_t)kk * 16 + (wave * 2 + nt)) * 32 + lane) * 16;
            v16bf bhi = *(const v16bf*)(whi + foff);   // 32B contiguous
            v16bf blo = *(const v16bf*)(wlo + foff);
            acc[nt] = __builtin_amdgcn_wmma_f32_16x16x32_bf16(
                          false, ahi, false, bhi, (short)0, acc[nt], false, false);
            acc[nt] = __builtin_amdgcn_wmma_f32_16x16x32_bf16(
                          false, ahi, false, blo, (short)0, acc[nt], false, false);
            acc[nt] = __builtin_amdgcn_wmma_f32_16x16x32_bf16(
                          false, alo, false, bhi, (short)0, acc[nt], false, false);
        }
    }

    // C layout: lanes 0-15 -> N=lane, M=r ; lanes 16-31 -> N=lane-16, M=r+8
    #pragma unroll
    for (int nt = 0; nt < 2; ++nt) {
        int n = wave * 32 + nt * 16 + am;
        float bv = bias[n];
        #pragma unroll
        for (int rr = 0; rr < 8; ++rr) {
            int m = rr + (g << 3);
            float v = acc[nt][rr] + bv;
            sH[m * 256 + n] = v > 0.0f ? v : 0.0f;   // ReLU
        }
    }
    __syncthreads();

    // LayerNorm over 256 channels, 16 threads per row
    const int row = threadIdx.x >> 4;
    const int j   = threadIdx.x & 15;
    {
        float s = 0.0f, s2 = 0.0f;
        for (int c = j; c < 256; c += 16) {
            float v = sH[row * 256 + c];
            s += v; s2 += v * v;
        }
        sRed[0][row][j] = s;
        sRed[1][row][j] = s2;
    }
    __syncthreads();
    if (j == 0) {
        float ts = 0.0f, ts2 = 0.0f;
        for (int q = 0; q < 16; ++q) { ts += sRed[0][row][q]; ts2 += sRed[1][row][q]; }
        float mu  = ts * (1.0f / 256.0f);
        float var = ts2 * (1.0f / 256.0f) - mu * mu;
        sMu[row] = mu;
        sRs[row] = rsqrtf(var + EPSc);
    }
    __syncthreads();

    if (mode == 0) {
        float* outB = hout + ((size_t)b * Lc + r0) * 256;
        for (int i = threadIdx.x; i < 16 * 256; i += 256) {
            int m = i >> 8, c = i & 255;
            outB[i] = (sH[i] - sMu[m]) * sRs[m] * gam[c] + bet[c];
        }
    } else {
        for (int i = threadIdx.x; i < 16 * 256; i += 256) {
            int m = i >> 8, c = i & 255;
            sH[i] = (sH[i] - sMu[m]) * sRs[m] * gam[c] + bet[c];
        }
        __syncthreads();
        float s = 0.0f;
        for (int c = j; c < 256; c += 16) s += sH[row * 256 + c] * lw[c];
        sRed[0][row][j] = s;
        __syncthreads();
        if (j == 0) {
            float t = lb[0];
            for (int q = 0; q < 16; ++q) t += sRed[0][row][q];
            int l = r0 + row;
            if (smask[b * Lc + l]) t = 0.0f;
            pred[b * Lc + l] = t;
        }
    }
}

// ---------------------------------------------------------------------------
// xe = x + conv1d(pitch,[9,1,H],pad4) + conv1d(energy,[9,1,H],pad4)
// ---------------------------------------------------------------------------
__global__ __launch_bounds__(256)
void embed_add_kernel(const float* __restrict__ x,
                      const float* __restrict__ pt,
                      const float* __restrict__ et,
                      const float* __restrict__ pw,   // (9,1,256)
                      const float* __restrict__ ew,   // (9,1,256)
                      float* __restrict__ xe)
{
    int bl = blockIdx.x;            // b*L + l
    int b = bl / Lc, l = bl % Lc;
    int h = threadIdx.x;
    float acc = x[(size_t)bl * 256 + h];
    #pragma unroll
    for (int k = 0; k < 9; ++k) {
        int lp = l + k - 4;
        if (lp >= 0 && lp < Lc) {
            float ps = pt[b * Lc + lp];
            float es = et[b * Lc + lp];
            acc += ps * pw[k * 256 + h] + es * ew[k * 256 + h];
        }
    }
    xe[(size_t)bl * 256 + h] = acc;
}

// ---------------------------------------------------------------------------
// per-batch inclusive cumsum of durations (Hillis-Steele over L=512)
// ---------------------------------------------------------------------------
__global__ __launch_bounds__(512)
void cumsum_kernel(const int* __restrict__ dur,
                   int* __restrict__ cum,
                   float* __restrict__ mel_len_out)
{
    __shared__ int s[512];
    int b = blockIdx.x, t = threadIdx.x;
    s[t] = dur[b * Lc + t];
    __syncthreads();
    for (int off = 1; off < Lc; off <<= 1) {
        int v = (t >= off) ? s[t - off] : 0;
        __syncthreads();
        s[t] += v;
        __syncthreads();
    }
    cum[b * Lc + t] = s[t];
    if (t == Lc - 1) mel_len_out[b] = (float)s[t];
}

// ---------------------------------------------------------------------------
// length regulate: out[b,t,:] = (t < mel_len) ? xe[b, idx(t), :] : 0
// idx(t) = min(searchsorted_right(cum[b], t), L-1).  8 t per block.
// Also mirrors mel_mask (bool -> float) into d_out.
// ---------------------------------------------------------------------------
static constexpr int TWc = 8;
__global__ __launch_bounds__(256)
void regulate_kernel(const float* __restrict__ xe,
                     const int* __restrict__ cum,
                     const unsigned char* __restrict__ melmask_in,
                     float* __restrict__ out,
                     float* __restrict__ maskout)
{
    __shared__ int sIdx[TWc];
    __shared__ int sValid[TWc];
    int b   = blockIdx.y;
    int t0  = blockIdx.x * TWc;
    int tid = threadIdx.x;
    const int* cb = cum + b * Lc;

    if (tid < TWc) {
        int t = t0 + tid;
        int lo = 0, hi = Lc;
        while (lo < hi) {                      // upper_bound
            int mid = (lo + hi) >> 1;
            if (cb[mid] <= t) lo = mid + 1; else hi = mid;
        }
        sIdx[tid]   = lo < Lc ? lo : Lc - 1;
        sValid[tid] = (t < cb[Lc - 1]);
        maskout[b * Tc + t] = melmask_in[b * Tc + t] ? 1.0f : 0.0f;
    }
    __syncthreads();

    #pragma unroll
    for (int i = 0; i < TWc; ++i) {
        int t = t0 + i;
        float v = sValid[i] ? xe[((size_t)b * Lc + sIdx[i]) * 256 + tid] : 0.0f;
        out[((size_t)b * Tc + t) * 256 + tid] = v;
    }
}

// ---------------------------------------------------------------------------
// launch
// ---------------------------------------------------------------------------
extern "C" void kernel_launch(void* const* d_in, const int* in_sizes, int n_in,
                              void* d_out, int out_size, void* d_ws, size_t ws_size,
                              hipStream_t stream)
{
    (void)in_sizes; (void)n_in; (void)out_size; (void)ws_size;

    // inputs, in setup_inputs() insertion order (nested param dicts in their
    // own insertion order: w1,b1,g1,be1,w2,b2,g2,be2,lw,lb)
    const float*         x        = (const float*)d_in[0];
    const unsigned char* src_mask = (const unsigned char*)d_in[1];
    const unsigned char* mel_mask = (const unsigned char*)d_in[2];
    const int*           dur_t    = (const int*)d_in[3];
    const float*         pitch_t  = (const float*)d_in[4];
    const float*         energy_t = (const float*)d_in[5];
    // d_in[6] = max_len scalar (== T, compile-time constant here)
    const int pbase[3] = {7, 17, 27};                    // dur, pitch, energy
    const float* pw = (const float*)d_in[37];
    const float* ew = (const float*)d_in[38];

    // d_out layout (floats, return order)
    float* out        = (float*)d_out;
    float* dur_pred   = out + (size_t)Bc * Tc * 256;
    float* pitch_pred = dur_pred + Bc * Lc;
    float* energy_pred= pitch_pred + Bc * Lc;
    float* mel_len    = energy_pred + Bc * Lc;
    float* mask_out   = mel_len + Bc;
    float* preds[3]   = {dur_pred, pitch_pred, energy_pred};

    // workspace: h1/xe buffer + cumsum ints + 6x (hi,lo) weight packs
    char*  wsb  = (char*)d_ws;
    float* h1   = (float*)wsb;
    int*   cum  = (int*)(wsb + H1_BYTES);
    __bf16* wpk = (__bf16*)(wsb + H1_BYTES + CUM_BYTES);
    // conv c in [0,6): hi at wpk + c*2*PACK_ELEMS, lo at +PACK_ELEMS

    const dim3 blk(256);
    const dim3 pgrid(24, 16);
    const int nblocks = Bc * (Lc / 16);

    // pre-split all 6 conv weights (tiny; L2-resident afterwards)
    for (int p = 0; p < 3; ++p) {
        const float* w1 = (const float*)d_in[pbase[p] + 0];
        const float* w2 = (const float*)d_in[pbase[p] + 4];
        __bf16* h1p = wpk + (size_t)(p * 2 + 0) * 2 * PACK_ELEMS;
        __bf16* h2p = wpk + (size_t)(p * 2 + 1) * 2 * PACK_ELEMS;
        pack_weights_kernel<<<pgrid, 32, 0, stream>>>(w1, h1p, h1p + PACK_ELEMS);
        pack_weights_kernel<<<pgrid, 32, 0, stream>>>(w2, h2p, h2p + PACK_ELEMS);
    }

    for (int p = 0; p < 3; ++p) {
        const float* b1  = (const float*)d_in[pbase[p] + 1];
        const float* g1  = (const float*)d_in[pbase[p] + 2];
        const float* be1 = (const float*)d_in[pbase[p] + 3];
        const float* b2  = (const float*)d_in[pbase[p] + 5];
        const float* g2  = (const float*)d_in[pbase[p] + 6];
        const float* be2 = (const float*)d_in[pbase[p] + 7];
        const float* lw  = (const float*)d_in[pbase[p] + 8];
        const float* lb  = (const float*)d_in[pbase[p] + 9];
        const __bf16* w1h = wpk + (size_t)(p * 2 + 0) * 2 * PACK_ELEMS;
        const __bf16* w2h = wpk + (size_t)(p * 2 + 1) * 2 * PACK_ELEMS;

        conv_ln_wmma_kernel<<<nblocks, blk, 0, stream>>>(
            x, w1h, w1h + PACK_ELEMS, b1, g1, be1,
            h1, nullptr, nullptr, nullptr, nullptr, 0);
        conv_ln_wmma_kernel<<<nblocks, blk, 0, stream>>>(
            h1, w2h, w2h + PACK_ELEMS, b2, g2, be2,
            nullptr, lw, lb, src_mask, preds[p], 1);
    }

    // xe overwrites h1 slot (all predictors done by stream order)
    embed_add_kernel<<<Bc * Lc, blk, 0, stream>>>(x, pitch_t, energy_t, pw, ew, h1);
    cumsum_kernel<<<Bc, 512, 0, stream>>>(dur_t, cum, mel_len);
    regulate_kernel<<<dim3(Tc / TWc, Bc), blk, 0, stream>>>(h1, cum, mel_mask, out, mask_out);
}
